// HeadAttention_84095459656221
// MI455X (gfx1250) — matrix operand order repaired and checked by reading, not compiled, not run on hardware
//
#include <hip/hip_runtime.h>
#include <hip/hip_bf16.h>

// Problem constants (match reference)
#define EMB   1024
#define HEADD 64
#define BATCH 4
#define SEQ   4096   // = 1<<12

typedef _Float16 v16h __attribute__((ext_vector_type(16)));
typedef _Float16 v8h  __attribute__((ext_vector_type(8)));
typedef _Float16 v4h  __attribute__((ext_vector_type(4)));
typedef float    v8f  __attribute__((ext_vector_type(8)));

static __device__ __forceinline__ v16h cat8(v8h lo, v8h hi) {
  return __builtin_shufflevector(lo, hi, 0,1,2,3,4,5,6,7,8,9,10,11,12,13,14,15);
}
static __device__ __forceinline__ v8f wmma16(v16h a, v16h b, v8f c) {
  // D = A(16x32 f16) * B(32x16 f16) + C(16x16 f32)
  return __builtin_amdgcn_wmma_f32_16x16x32_f16(false, a, false, b, (short)0, c,
                                                false, false);
}
static __device__ __forceinline__ float fast_exp2(float x) {
#if __has_builtin(__builtin_amdgcn_exp2f)
  return __builtin_amdgcn_exp2f(x);   // single v_exp_f32
#else
  return exp2f(x);
#endif
}

// ---------------------------------------------------------------------------
// Kernel 1: fused QKV projection.
//   One block = one 16-row tile of x (read once, staged to LDS as f16).
//   12 waves: wave w -> matrix (w/4 in {Q,K,V}) x column chunk (w%4 -> 16 cols).
//   Q,K stored row-major f16 [B*S][64]; V stored transposed f16 [B][64][S]
//   so the attention kernel's B-operand loads are contiguous.
// ---------------------------------------------------------------------------
__global__ __launch_bounds__(384)
void qkv_proj_kernel(const float* __restrict__ x,
                     const float* __restrict__ Wq, const float* __restrict__ bq,
                     const float* __restrict__ Wk, const float* __restrict__ bk,
                     const float* __restrict__ Wv, const float* __restrict__ bv,
                     _Float16* __restrict__ Qh, _Float16* __restrict__ Kh,
                     _Float16* __restrict__ Vt)
{
  __shared__ _Float16 xs[16 * 32];           // 16 rows x 32 k of x, f16

  const int tid   = threadIdx.x;
  const int lane  = tid & 31;
  const int wave  = tid >> 5;                // 0..11
  const int widx  = wave >> 2;               // 0=Q 1=K 2=V
  const int chunk = wave & 3;                // output column chunk
  const int col16 = lane & 15;
  const int col   = chunk * 16 + col16;      // output column 0..63
  const int kb2   = (lane < 16) ? 0 : 8;     // A-layout K split
  const int kbB   = (lane < 16) ? 0 : 16;    // B-layout K split
  const int row0  = blockIdx.x * 16;         // global row in [0, B*S)

  const float* W    = (widx == 0) ? Wq : (widx == 1) ? Wk : Wv;
  const float* bias = (widx == 0) ? bq : (widx == 1) ? bk : bv;

  v8f acc = {};

  for (int k0 = 0; k0 < EMB; k0 += 32) {
    // Stage x[row0..row0+15][k0..k0+31] -> LDS (f32 -> f16)
    if (tid < 128) {
      int idx = tid * 4;                     // element index in 16x32 tile
      int r = idx >> 5, c = idx & 31;
      float4 xv = *(const float4*)(x + (size_t)(row0 + r) * EMB + k0 + c);
      v4h hv = { (_Float16)xv.x, (_Float16)xv.y, (_Float16)xv.z, (_Float16)xv.w };
      *(v4h*)(&xs[idx]) = hv;
    }
    __syncthreads();

    // A operand from LDS (row = lane%16, k [kbB..kbB+15] contiguous)
    v8h a_lo = *(const v8h*)(&xs[col16 * 32 + kbB]);
    v8h a_hi = *(const v8h*)(&xs[col16 * 32 + kbB + 8]);
    v16h A = cat8(a_lo, a_hi);

    // B operand: W[k0+kbB .. +15][col] strided gather (L2-resident), f32->f16
    v16h Bm;
    #pragma unroll
    for (int j = 0; j < 16; ++j)
      Bm[j] = (_Float16)W[(size_t)(k0 + kbB + j) * HEADD + col];

    acc = wmma16(A, Bm, acc);
    __syncthreads();
  }

  // Add bias, convert, store.
  const float bsc  = bias[col];
  const int   roff = (lane < 16) ? 0 : 8;
  #pragma unroll
  for (int g = 0; g < 8; ++g) {
    const int m = row0 + g + roff;           // global row in [0, B*S)
    const _Float16 hv = (_Float16)(acc[g] + bsc);
    if (widx == 0) {
      Qh[(size_t)m * HEADD + col] = hv;
    } else if (widx == 1) {
      Kh[(size_t)m * HEADD + col] = hv;
    } else {
      const int b = m >> 12, s = m & (SEQ - 1);
      Vt[(((size_t)b * HEADD + col) << 12) + s] = hv;   // V transposed [B][64][S]
    }
  }
}

// ---------------------------------------------------------------------------
// Kernel 2: causal flash attention. One wave per (batch, 16-query tile).
//   32-key blocks: 4 WMMA scores (K-dim 64 = 2x32, two 16-key tiles),
//   base-2 online softmax with xor-shuffle reductions confined to the
//   16-lane halves of the wave32 C-layout, P transposed via LDS with an
//   explicit s_wait_dscnt, 4 WMMA for P@V on transposed V.
//   Main loop is unmasked; exactly one masked diagonal block at the end.
// ---------------------------------------------------------------------------

// One 32-key block. DOMASK is a literal 0/1 so the branch folds away.
#define PROCESS_BLOCK(KBv, DOMASK)                                             \
  {                                                                            \
    const int kb = (KBv);                                                      \
    const _Float16* Krow0 = Kbase + (size_t)(kb + col16) * HEADD;              \
    const _Float16* Krow1 = Krow0 + 16 * HEADD;                                \
    /* speculative prefetch of next block (OOB silently dropped) */            \
    __builtin_prefetch(Krow0 + 32 * HEADD, 0, 0);                              \
    __builtin_prefetch(Krow1 + 32 * HEADD, 0, 0);                              \
    __builtin_prefetch(Vbase + (((size_t)col16) << 12) + kb + 32, 0, 0);       \
    __builtin_prefetch(Vbase + (((size_t)(48 + col16)) << 12) + kb + 32, 0, 0);\
    v16h B00 = cat8(*(const v8h*)(Krow0 + kbB),                                \
                    *(const v8h*)(Krow0 + kbB + 8));                           \
    v16h B01 = cat8(*(const v8h*)(Krow0 + 32 + kbB),                           \
                    *(const v8h*)(Krow0 + 32 + kbB + 8));                      \
    v16h B10 = cat8(*(const v8h*)(Krow1 + kbB),                                \
                    *(const v8h*)(Krow1 + kbB + 8));                           \
    v16h B11 = cat8(*(const v8h*)(Krow1 + 32 + kbB),                           \
                    *(const v8h*)(Krow1 + 32 + kbB + 8));                      \
    v8f S0 = {}, S1 = {};                                                      \
    S0 = wmma16(A0, B00, S0);                                                  \
    S0 = wmma16(A1, B01, S0);                                                  \
    S1 = wmma16(A0, B10, S1);                                                  \
    S1 = wmma16(A1, B11, S1);                                                  \
    float mcur[8];                                                             \
    _Pragma("unroll")                                                          \
    for (int g = 0; g < 8; ++g) {                                              \
      float a = S0[g] * scale2;        /* base-2 domain */                     \
      float c = S1[g] * scale2;                                                \
      if (DOMASK) {                                                            \
        const int q = qt * 16 + g + roff;                                      \
        a = (kb + col16      <= q) ? a : NEG_INF;                              \
        c = (kb + 16 + col16 <= q) ? c : NEG_INF;                              \
      }                                                                        \
      S0[g] = a; S1[g] = c;                                                    \
      mcur[g] = fmaxf(a, c);                                                   \
    }                                                                          \
    _Pragma("unroll")                                                          \
    for (int msk = 1; msk <= 8; msk <<= 1) {                                   \
      _Pragma("unroll")                                                        \
      for (int g = 0; g < 8; ++g)                                              \
        mcur[g] = fmaxf(mcur[g], __shfl_xor(mcur[g], msk, 32));                \
    }                                                                          \
    float alpha[8], rsum[8];                                                   \
    _Pragma("unroll")                                                          \
    for (int g = 0; g < 8; ++g) {                                              \
      const float mn = fmaxf(m_run[g], mcur[g]);                               \
      alpha[g] = fast_exp2(m_run[g] - mn);                                     \
      m_run[g] = mn;                                                           \
      const float p0 = fast_exp2(S0[g] - mn);                                  \
      const float p1 = fast_exp2(S1[g] - mn);                                  \
      S0[g] = p0; S1[g] = p1;                                                  \
      rsum[g] = p0 + p1;                                                       \
    }                                                                          \
    _Pragma("unroll")                                                          \
    for (int msk = 1; msk <= 8; msk <<= 1) {                                   \
      _Pragma("unroll")                                                        \
      for (int g = 0; g < 8; ++g)                                              \
        rsum[g] += __shfl_xor(rsum[g], msk, 32);                               \
    }                                                                          \
    _Pragma("unroll")                                                          \
    for (int g = 0; g < 8; ++g) {                                              \
      l_run[g] = l_run[g] * alpha[g] + rsum[g];                                \
      O0[g] *= alpha[g]; O1[g] *= alpha[g];                                    \
      O2[g] *= alpha[g]; O3[g] *= alpha[g];                                    \
    }                                                                          \
    /* P: C-layout -> LDS -> A-layout (wave-local transpose) */                \
    _Pragma("unroll")                                                          \
    for (int g = 0; g < 8; ++g) {                                              \
      const int row = g + roff;                                                \
      P[row * 32 + col16]      = (_Float16)S0[g];                              \
      P[row * 32 + 16 + col16] = (_Float16)S1[g];                              \
    }                                                                          \
    asm volatile("s_wait_dscnt 0x0" ::: "memory");                             \
    v16h Ap;                                                                   \
    {                                                                          \
      v8h lo = *(const v8h*)(P + col16 * 32 + kbB);                            \
      v8h hi = *(const v8h*)(P + col16 * 32 + 8 + kbB);                        \
      Ap = cat8(lo, hi);                                                       \
    }                                                                          \
    /* O += P @ V  (V transposed: contiguous 32B B-operand loads) */           \
    {                                                                          \
      const _Float16* vp = VbaseL + (((size_t)(0 * 16 + col16)) << 12) + kb;   \
      O0 = wmma16(Ap, cat8(*(const v8h*)vp, *(const v8h*)(vp + 8)), O0);       \
    }                                                                          \
    {                                                                          \
      const _Float16* vp = VbaseL + (((size_t)(1 * 16 + col16)) << 12) + kb;   \
      O1 = wmma16(Ap, cat8(*(const v8h*)vp, *(const v8h*)(vp + 8)), O1);       \
    }                                                                          \
    {                                                                          \
      const _Float16* vp = VbaseL + (((size_t)(2 * 16 + col16)) << 12) + kb;   \
      O2 = wmma16(Ap, cat8(*(const v8h*)vp, *(const v8h*)(vp + 8)), O2);       \
    }                                                                          \
    {                                                                          \
      const _Float16* vp = VbaseL + (((size_t)(3 * 16 + col16)) << 12) + kb;   \
      O3 = wmma16(Ap, cat8(*(const v8h*)vp, *(const v8h*)(vp + 8)), O3);       \
    }                                                                          \
  }

__global__ __launch_bounds__(256)
void flash_attn_kernel(const _Float16* __restrict__ Qh,
                       const _Float16* __restrict__ Kh,
                       const _Float16* __restrict__ Vt,
                       float* __restrict__ out)
{
  __shared__ _Float16 Pbuf[8][16 * 32];      // per-wave P scratch (1 KB each)

  const int tid   = threadIdx.x;
  const int lane  = tid & 31;
  const int wave  = tid >> 5;
  const int wgid  = blockIdx.x * 8 + wave;   // 0..1023
  const int b     = wgid >> 8;               // batch
  const int qt    = wgid & 255;              // query tile
  const int col16 = lane & 15;
  const int roff  = (lane < 16) ? 0 : 8;     // C-layout row split
  const int kb2   = (lane < 16) ? 0 : 8;     // A-layout K split
  const int kbB   = (lane < 16) ? 0 : 16;    // B-layout K split
  _Float16* P = &Pbuf[wave][0];

  const float NEG_INF = -__builtin_inff();
  // base-2 softmax: fold 1/sqrt(64) and log2(e) into one score scale
  const float scale2 = 0.125f * 1.44269504088896340736f;

  const _Float16* Kbase  = Kh + (size_t)(b * SEQ) * HEADD;
  const _Float16* Vbase  = Vt + (((size_t)b * HEADD) << 12);
  const _Float16* VbaseL = Vbase + kbB;      // lane's K-split within V blocks

  // Q tile A operands (K-dim 64 -> two 16x32 A tiles), loaded once.
  const _Float16* Qrow = Qh + ((size_t)(b * SEQ) + qt * 16 + col16) * HEADD;
  v16h A0, A1;
  {
    v8h lo = *(const v8h*)(Qrow + kbB);
    v8h hi = *(const v8h*)(Qrow + kbB + 8);
    A0 = cat8(lo, hi);
    lo = *(const v8h*)(Qrow + 32 + kbB);
    hi = *(const v8h*)(Qrow + 32 + kbB + 8);
    A1 = cat8(lo, hi);
  }

  v8f O0 = {}, O1 = {}, O2 = {}, O3 = {};
  float m_run[8], l_run[8];
  #pragma unroll
  for (int g = 0; g < 8; ++g) { m_run[g] = NEG_INF; l_run[g] = 0.0f; }

  // Blocks [0, nfull) need no causal mask; exactly one masked diagonal block.
  const int nfull = (qt * 16 + 1) >> 5;
  for (int it = 0; it < nfull; ++it) {
    PROCESS_BLOCK(it * 32, 0)
  }
  PROCESS_BLOCK(nfull * 32, 1)

  // ---- normalize and store f32 output ----
  #pragma unroll
  for (int g = 0; g < 8; ++g) {
    const float inv = 1.0f / l_run[g];
    const int m = b * SEQ + qt * 16 + g + roff;
    float* orow = out + (size_t)m * HEADD;
    orow[col16]      = O0[g] * inv;
    orow[16 + col16] = O1[g] * inv;
    orow[32 + col16] = O2[g] * inv;
    orow[48 + col16] = O3[g] * inv;
  }
}

// ---------------------------------------------------------------------------
extern "C" void kernel_launch(void* const* d_in, const int* in_sizes, int n_in,
                              void* d_out, int out_size, void* d_ws, size_t ws_size,
                              hipStream_t stream)
{
  const float* x  = (const float*)d_in[0];
  const float* Wq = (const float*)d_in[1];
  const float* bq = (const float*)d_in[2];
  const float* Wk = (const float*)d_in[3];
  const float* bk = (const float*)d_in[4];
  const float* Wv = (const float*)d_in[5];
  const float* bv = (const float*)d_in[6];
  float* out = (float*)d_out;

  // Workspace: Qh | Kh | Vt, each B*S*64 f16 = 2 MB (6 MB total).
  const size_t QK_BYTES = (size_t)BATCH * SEQ * HEADD * sizeof(_Float16);
  char* ws = (char*)d_ws;
  _Float16* Qh = (_Float16*)(ws);
  _Float16* Kh = (_Float16*)(ws + QK_BYTES);
  _Float16* Vt = (_Float16*)(ws + 2 * QK_BYTES);

  // 1024 row tiles of 16; 12 waves (Q,K,V x 4 column chunks) per block.
  qkv_proj_kernel<<<(BATCH * SEQ) / 16, 384, 0, stream>>>(
      x, Wq, bq, Wk, bk, Wv, bv, Qh, Kh, Vt);

  // 1024 (batch, q-tile) waves, 8 waves per block.
  flash_attn_kernel<<<(BATCH * (SEQ / 16)) / 8, 256, 0, stream>>>(
      Qh, Kh, Vt, out);
}
